// SNNDensityNet_17858474017375
// MI455X (gfx1250) — compile-verified, matrix operand order checked by reading, not run on hardware
//
#include <hip/hip_runtime.h>
#include <hip/hip_bf16.h>
#include <math.h>

typedef __attribute__((ext_vector_type(16))) __bf16 v16bf;
typedef __attribute__((ext_vector_type(8)))  float  v8f;
typedef __attribute__((ext_vector_type(4)))  unsigned int v4u;
typedef __attribute__((ext_vector_type(8)))  unsigned int v8u;

#define DDIM   768
#define NROWS  8192
#define BROWS  4096
#define PROWS  2048
#define NCLASS 256
#define KN     20
#define TAUF   0.07f
#define KCHUNKS (DDIM / 32)   // 24 WMMA k-steps

// d_out float offsets: (preds, densities, knn_indices, peak_mask)
#define DENS_OFF  (BROWS * NCLASS)
#define KNN_OFF   (DENS_OFF + NROWS)
#define MASK_OFF  (KNN_OFF + NROWS * KN)

// ---- TDM: stage a (tile_rows x 768) bf16 row-tile from global into LDS ----
// D# per CDNA5 ISA: group0 = count/lds_addr/global_addr/type,
// group1 = data_size + tensor/tile dims + stride. 2D tile -> 2 SGPR groups.
__device__ __forceinline__ void tdm_load_tile(const void* gsrc, void* lds_dst,
                                              unsigned tile_rows,
                                              unsigned tensor_rows)
{
    unsigned long long ga = (unsigned long long)(size_t)gsrc;
    v4u g0;
    g0[0] = 1u;                                    // count=1, user descriptor
    g0[1] = (unsigned)(size_t)lds_dst;             // LDS byte offset (addr[31:0])
    g0[2] = (unsigned)(ga & 0xFFFFFFFFull);        // global_addr[31:0]
    g0[3] = (unsigned)((ga >> 32) & 0x1FFFFFFull)  // global_addr[56:32]
          | (2u << 30);                            // type=2 ("image")
    v8u g1;
    g1[0] = 0x00010000u;                           // wg_mask=0, data_size=2B
    g1[1] = ((unsigned)DDIM & 0xFFFFu) << 16;      // tensor_dim0[15:0]
    g1[2] = (((unsigned)DDIM >> 16) & 0xFFFFu)     // tensor_dim0[31:16]
          | ((tensor_rows & 0xFFFFu) << 16);       // tensor_dim1[15:0]
    g1[3] = ((tensor_rows >> 16) & 0xFFFFu)        // tensor_dim1[31:16]
          | (((unsigned)DDIM & 0xFFFFu) << 16);    // tile_dim0 = 768
    g1[4] = tile_rows & 0xFFFFu;                   // tile_dim1 (tile_dim2=0)
    g1[5] = (unsigned)DDIM;                        // tensor_dim0_stride[31:0]
    g1[6] = 0u;
    g1[7] = 0u;
    asm volatile("tensor_load_to_lds %0, %1" :: "s"(g0), "s"(g1) : "memory");
}

// ---------------- row L2-normalize fp32 -> bf16 (one wave per row) ----------
__global__ void __launch_bounds__(32) row_normalize_bf16(
    const float* __restrict__ in, __bf16* __restrict__ out)
{
    const int row  = blockIdx.x;
    const int lane = threadIdx.x;
    const float* src = in + (size_t)row * DDIM;
    float ss = 0.f;
#pragma unroll
    for (int t = 0; t < DDIM / 32; ++t) {
        float v = src[lane + t * 32];
        ss += v * v;
    }
#pragma unroll
    for (int m = 16; m >= 1; m >>= 1) ss += __shfl_xor(ss, m, 32);
    const float inv = rsqrtf(ss);
    __bf16* dst = out + (size_t)row * DDIM;
#pragma unroll
    for (int t = 0; t < DDIM / 32; ++t)
        dst[lane + t * 32] = (__bf16)(src[lane + t * 32] * inv);
}

// --- fused sim = f@f^T (32x32 tile per wave, bf16 WMMA) + streaming top-21 --
__global__ void __launch_bounds__(32) sim_topk_kernel(
    const __bf16* __restrict__ fnorm,
    float* __restrict__ dens_out,
    float* __restrict__ knn_out_f,
    int*   __restrict__ knn_out_i)
{
    __shared__ __bf16 ldsA[32 * DDIM];   // 48 KB: this wave's 32 A-rows
    __shared__ float  stage[32 * 32];    // 4 KB: one 32x32 sim block
    __shared__ float  topV[32 * 21];
    __shared__ int    topI[32 * 21];

    const int lane = threadIdx.x;
    const int m0   = blockIdx.x * 32;

    tdm_load_tile(fnorm + (size_t)m0 * DDIM, ldsA, 32, NROWS);
    for (int t = 0; t < 21; ++t) {
        topV[lane * 21 + t] = -1e30f;
        topI[lane * 21 + t] = -1;
    }
    __builtin_amdgcn_s_wait_tensorcnt(0);
    __syncthreads();

    const int rsel = lane & 15;
    const int koff = (lane >> 4) << 4;   // 0 or 16: this lane's K half
    const __bf16* arow0 = ldsA + rsel * DDIM + koff;
    const __bf16* arow1 = ldsA + (16 + rsel) * DDIM + koff;

    for (int nt = 0; nt < NROWS / 32; ++nt) {
        const int n0 = nt * 32;
        const __bf16* brow0 = fnorm + (size_t)(n0 + rsel) * DDIM + koff;
        const __bf16* brow1 = fnorm + (size_t)(n0 + 16 + rsel) * DDIM + koff;
        if (nt + 1 < NROWS / 32)
            __builtin_prefetch(fnorm + (size_t)(n0 + 32 + rsel) * DDIM, 0, 3);

        v8f c00 = {}, c01 = {}, c10 = {}, c11 = {};
#pragma unroll 2
        for (int kc = 0; kc < KCHUNKS; ++kc) {
            v16bf a0 = *(const v16bf*)(arow0 + kc * 32);
            v16bf a1 = *(const v16bf*)(arow1 + kc * 32);
            v16bf b0 = *(const v16bf*)(brow0 + kc * 32);
            v16bf b1 = *(const v16bf*)(brow1 + kc * 32);
            c00 = __builtin_amdgcn_wmma_f32_16x16x32_bf16(false, a0, false, b0, (short)0, c00, false, false);
            c01 = __builtin_amdgcn_wmma_f32_16x16x32_bf16(false, a0, false, b1, (short)0, c01, false, false);
            c10 = __builtin_amdgcn_wmma_f32_16x16x32_bf16(false, a1, false, b0, (short)0, c10, false, false);
            c11 = __builtin_amdgcn_wmma_f32_16x16x32_bf16(false, a1, false, b1, (short)0, c11, false, false);
        }
        // D layout: VGPR r -> (lanes 0-15: M=r, lanes 16-31: M=r+8), N=lane%16
#pragma unroll
        for (int r = 0; r < 8; ++r) {
            int rr = r + ((lane < 16) ? 0 : 8);
            stage[rr * 32 + rsel]             = c00[r];
            stage[rr * 32 + 16 + rsel]        = c01[r];
            stage[(16 + rr) * 32 + rsel]      = c10[r];
            stage[(16 + rr) * 32 + 16 + rsel] = c11[r];
        }
        __syncthreads();
        // every lane owns one row of the 32-row tile
        {
            float* tv = topV + lane * 21;
            int*   ti = topI + lane * 21;
            for (int j = 0; j < 32; ++j) {
                float v = stage[lane * 32 + j];
                if (v > tv[20]) {
                    int i = 20;
                    while (i > 0 && v > tv[i - 1]) {
                        tv[i] = tv[i - 1]; ti[i] = ti[i - 1]; --i;
                    }
                    tv[i] = v; ti[i] = n0 + j;
                }
            }
        }
        __syncthreads();
    }

    {
        const int row = m0 + lane;
        float s = 0.f;
        for (int t = 1; t <= KN; ++t) s += topV[lane * 21 + t]; // rank 0 = self
        dens_out[row] = s / (float)KN;
        for (int t = 0; t < KN; ++t) {
            int idx = topI[lane * 21 + t + 1];
            knn_out_i[(size_t)row * KN + t] = idx;
            knn_out_f[(size_t)row * KN + t] = (float)idx;
        }
    }
}

// ---------------- peak mask: all(dens[i] > dens[knn[i,:]]) ------------------
__global__ void peak_mask_kernel(const float* __restrict__ dens,
                                 const int* __restrict__ knn,
                                 float* __restrict__ mask_out)
{
    int i = blockIdx.x * blockDim.x + threadIdx.x;
    if (i >= NROWS) return;
    float di = dens[i];
    bool ok = true;
#pragma unroll
    for (int j = 0; j < KN; ++j)
        ok = ok && (di > dens[knn[(size_t)i * KN + j]]);
    mask_out[i] = ok ? 1.f : 0.f;
}

// -- logits = hn@pn^T/tau (16x32 WMMA tile) + online softmax + label scatter -
__global__ void __launch_bounds__(32) preds_kernel(
    const __bf16* __restrict__ hn, const __bf16* __restrict__ pn,
    const int* __restrict__ labels, float* __restrict__ preds_out)
{
    __shared__ __bf16 ldsA[16 * DDIM];       // 24 KB
    __shared__ float  stage[16 * 32];        // 2 KB
    __shared__ int    ldsLab[PROWS];         // 8 KB
    __shared__ float  predsL[16 * NCLASS];   // 16 KB

    const int lane = threadIdx.x;
    const int m0   = blockIdx.x * 16;

    tdm_load_tile(hn + (size_t)m0 * DDIM, ldsA, 16, BROWS);
    for (int t = lane; t < PROWS; t += 32) ldsLab[t] = labels[t];
    for (int t = lane; t < 16 * NCLASS; t += 32) predsL[t] = 0.f;
    __builtin_amdgcn_s_wait_tensorcnt(0);
    __syncthreads();

    const int rsel  = lane & 15;
    const int koff  = (lane >> 4) << 4;
    const int chalf = (lane >> 4) * 16;      // column half this lane reduces
    const __bf16* arow = ldsA + rsel * DDIM + koff;

    float m = -1e30f, s = 0.f;               // per-(row, col-half) online state

    // pass 1: online row max + sumexp, 32 columns per tile
    for (int nt = 0; nt < PROWS / 32; ++nt) {
        const int n0 = nt * 32;
        const __bf16* brow0 = pn + (size_t)(n0 + rsel) * DDIM + koff;
        const __bf16* brow1 = pn + (size_t)(n0 + 16 + rsel) * DDIM + koff;
        v8f c0 = {}, c1 = {};
#pragma unroll 2
        for (int kc = 0; kc < KCHUNKS; ++kc) {
            v16bf a  = *(const v16bf*)(arow + kc * 32);
            v16bf b0 = *(const v16bf*)(brow0 + kc * 32);
            v16bf b1 = *(const v16bf*)(brow1 + kc * 32);
            c0 = __builtin_amdgcn_wmma_f32_16x16x32_bf16(false, a, false, b0, (short)0, c0, false, false);
            c1 = __builtin_amdgcn_wmma_f32_16x16x32_bf16(false, a, false, b1, (short)0, c1, false, false);
        }
#pragma unroll
        for (int r = 0; r < 8; ++r) {
            int rr = r + ((lane < 16) ? 0 : 8);
            stage[rr * 32 + rsel]      = c0[r];
            stage[rr * 32 + 16 + rsel] = c1[r];
        }
        __syncthreads();
        const float* sp = stage + rsel * 32 + chalf;
        for (int j = 0; j < 16; ++j) {
            float l  = sp[j] * (1.0f / TAUF);
            float mn = fmaxf(m, l);
            s = s * __expf(m - mn) + __expf(l - mn);
            m = mn;
        }
        __syncthreads();
    }
    // merge the two column-half states for each row
    {
        float mo = __shfl_xor(m, 16, 32);
        float so = __shfl_xor(s, 16, 32);
        float mt = fmaxf(m, mo);
        s = s * __expf(m - mt) + so * __expf(mo - mt);
        m = mt;
    }

    // pass 2: recompute logits, scatter probs by label (halves serialized)
    for (int nt = 0; nt < PROWS / 32; ++nt) {
        const int n0 = nt * 32;
        const __bf16* brow0 = pn + (size_t)(n0 + rsel) * DDIM + koff;
        const __bf16* brow1 = pn + (size_t)(n0 + 16 + rsel) * DDIM + koff;
        v8f c0 = {}, c1 = {};
#pragma unroll 2
        for (int kc = 0; kc < KCHUNKS; ++kc) {
            v16bf a  = *(const v16bf*)(arow + kc * 32);
            v16bf b0 = *(const v16bf*)(brow0 + kc * 32);
            v16bf b1 = *(const v16bf*)(brow1 + kc * 32);
            c0 = __builtin_amdgcn_wmma_f32_16x16x32_bf16(false, a, false, b0, (short)0, c0, false, false);
            c1 = __builtin_amdgcn_wmma_f32_16x16x32_bf16(false, a, false, b1, (short)0, c1, false, false);
        }
#pragma unroll
        for (int r = 0; r < 8; ++r) {
            int rr = r + ((lane < 16) ? 0 : 8);
            stage[rr * 32 + rsel]      = c0[r];
            stage[rr * 32 + 16 + rsel] = c1[r];
        }
        __syncthreads();
        const float invs = 1.0f / s;
        if (lane < 16) {                      // phase A: cols 0..15
            for (int j = 0; j < 16; ++j) {
                float l = stage[rsel * 32 + j] * (1.0f / TAUF);
                predsL[rsel * NCLASS + ldsLab[n0 + j]] += __expf(l - m) * invs;
            }
        }
        __syncthreads();
        if (lane >= 16) {                     // phase B: cols 16..31
            for (int j = 0; j < 16; ++j) {
                float l = stage[rsel * 32 + 16 + j] * (1.0f / TAUF);
                predsL[rsel * NCLASS + ldsLab[n0 + 16 + j]] += __expf(l - m) * invs;
            }
        }
        __syncthreads();
    }

    float* dst = preds_out + (size_t)m0 * NCLASS;
    for (int t = lane; t < 16 * NCLASS; t += 32) dst[t] = predsL[t];
}

extern "C" void kernel_launch(void* const* d_in, const int* in_sizes, int n_in,
                              void* d_out, int out_size, void* d_ws, size_t ws_size,
                              hipStream_t stream) {
    const float* features = (const float*)d_in[0];
    const float* h_i      = (const float*)d_in[1];
    const float* peaks    = (const float*)d_in[2];
    const int*   labels   = (const int*)d_in[3];

    char* ws = (char*)d_ws;
    __bf16* fnorm = (__bf16*)ws;                                          // 12 MB
    __bf16* hnn   = (__bf16*)(ws + (size_t)NROWS * DDIM * 2);             //  6 MB
    __bf16* pnn   = (__bf16*)(ws + (size_t)(NROWS + BROWS) * DDIM * 2);   //  3 MB
    int*    knnI  = (int*)(ws + (size_t)(NROWS + BROWS + PROWS) * DDIM * 2);

    float* out   = (float*)d_out;
    float* preds = out;
    float* dens  = out + DENS_OFF;
    float* knnF  = out + KNN_OFF;
    float* mask  = out + MASK_OFF;

    row_normalize_bf16<<<NROWS, 32, 0, stream>>>(features, fnorm);
    row_normalize_bf16<<<BROWS, 32, 0, stream>>>(h_i, hnn);
    row_normalize_bf16<<<PROWS, 32, 0, stream>>>(peaks, pnn);

    sim_topk_kernel<<<NROWS / 32, 32, 0, stream>>>(fnorm, dens, knnF, knnI);
    peak_mask_kernel<<<NROWS / 256, 256, 0, stream>>>(dens, knnI, mask);
    preds_kernel<<<BROWS / 16, 32, 0, stream>>>(hnn, pnn, labels, preds);
}
